// HealthMoELayer_12481174962385
// MI455X (gfx1250) — compile-verified, matrix operand order checked by reading, not run on hardware
//
#include <hip/hip_runtime.h>
#include <math.h>

// Problem constants (from reference)
#define E_    12
#define KTOP  3
#define B_    2
#define S_    2048
#define H_    2048
#define I_    5120
#define NTOK  (B_*S_)          // 4096 tokens
#define MAXTOK NTOK
#define TOTASSIGN (KTOP*NTOK)  // 12288 (token,expert) assignments

// GEMM tiling: 256 threads = 8 wave32; block tile 128x128x32,
// each wave owns a 32x64 subtile = 2x4 WMMA fragments.
#define BM 128
#define BN 128
#define BK 32

typedef __bf16 bf16;
typedef __attribute__((ext_vector_type(8)))  __bf16 v8bf;
typedef __attribute__((ext_vector_type(16))) __bf16 v16bf;
typedef __attribute__((ext_vector_type(8)))  float  v8f;
typedef __attribute__((ext_vector_type(4)))  unsigned int v4u;
typedef __attribute__((ext_vector_type(8)))  int v8i_;
typedef __attribute__((ext_vector_type(4)))  int v4i_;

// Load a 16-element bf16 WMMA fragment from LDS.
// CDNA5 16-bit A 16x32 layout: lane half selects K-subrange; elements 0..7 are
// K = half*8 + 0..7, elements 8..15 are K = 16 + half*8 + 0..7.
// `p` points at row_base + half*8 (16B aligned; row stride 40 bf16 = 80B).
__device__ __forceinline__ v16bf load_frag(const bf16* p) {
  v8bf lo = *(const v8bf*)(p);
  v8bf hi = *(const v8bf*)(p + 16);
  return __builtin_shufflevector(lo, hi, 0,1,2,3,4,5,6,7,8,9,10,11,12,13,14,15);
}

// ---------------------------------------------------------------------------
// 1) Router: logits -> softmax -> top3 -> renorm weights; atomic-append token
//    to its experts' lists; accumulate per-expert weight sums and counts.
// ---------------------------------------------------------------------------
__global__ __launch_bounds__(256) void router_kernel(
    const float* __restrict__ x, const float* __restrict__ Wr,
    const float* __restrict__ br,
    int* __restrict__ cnt, float* __restrict__ sumw, int* __restrict__ lists) {
  const int token = blockIdx.x;
  const int tid = threadIdx.x;
  float acc[E_];
#pragma unroll
  for (int e = 0; e < E_; ++e) acc[e] = 0.f;
  const float* xr = x + (size_t)token * H_;
  for (int h = tid; h < H_; h += 256) {
    float xv = xr[h];
    const float* wr = Wr + (size_t)h * E_;
#pragma unroll
    for (int e = 0; e < E_; ++e) acc[e] += xv * wr[e];
  }
  __shared__ float red[256];
  __shared__ float logits[E_];
  for (int e = 0; e < E_; ++e) {
    red[tid] = acc[e];
    __syncthreads();
    for (int off = 128; off > 0; off >>= 1) {
      if (tid < off) red[tid] += red[tid + off];
      __syncthreads();
    }
    if (tid == 0) logits[e] = red[0] + br[e];
    __syncthreads();
  }
  if (tid == 0) {
    float mx = logits[0];
#pragma unroll
    for (int e = 1; e < E_; ++e) mx = fmaxf(mx, logits[e]);
    float p[E_], s = 0.f;
#pragma unroll
    for (int e = 0; e < E_; ++e) { p[e] = __expf(logits[e] - mx); s += p[e]; }
    float inv = 1.f / s;
#pragma unroll
    for (int e = 0; e < E_; ++e) p[e] *= inv;
    int ti[KTOP]; float tv[KTOP];
    bool used[E_] = {false};
#pragma unroll
    for (int k = 0; k < KTOP; ++k) {
      float best = -1.f; int bi = 0;
      for (int e = 0; e < E_; ++e)
        if (!used[e] && p[e] > best) { best = p[e]; bi = e; }
      used[bi] = true; ti[k] = bi; tv[k] = best;
    }
    float m2 = fmaxf(tv[0], fmaxf(tv[1], tv[2]));
    float ew[KTOP], es = 0.f;
#pragma unroll
    for (int k = 0; k < KTOP; ++k) { ew[k] = __expf(tv[k] - m2); es += ew[k]; }
    float ies = 1.f / es;
#pragma unroll
    for (int k = 0; k < KTOP; ++k) {
      int e = ti[k];
      atomicAdd(&sumw[e], ew[k] * ies);
      int pos = atomicAdd(&cnt[e], 1);
      lists[e * MAXTOK + pos] = token;
    }
  }
}

// ---------------------------------------------------------------------------
// 2) Finalize: per-expert scalar w_e = sum/max(cnt,1); CSR base offsets.
// ---------------------------------------------------------------------------
__global__ void finalize_kernel(const int* __restrict__ cnt,
                                const float* __restrict__ sumw,
                                float* __restrict__ wexp, int* __restrict__ base) {
  int t = threadIdx.x;
  if (t < E_) wexp[t] = sumw[t] / fmaxf((float)cnt[t], 1.0f);
  __syncthreads();
  if (t == 0) {
    int b = 0;
    for (int e = 0; e < E_; ++e) { base[e] = b; b += cnt[e]; }
  }
}

// ---------------------------------------------------------------------------
// 3) Expert layer 1: h = relu(gather(x) @ w1[e] + b1[e]) stored bf16 in ws.
//    A rows gathered via token list (fp32 -> bf16), B transposed into LDS.
// ---------------------------------------------------------------------------
__global__ __launch_bounds__(256) void expert_ffn1(
    const float* __restrict__ x, const float* __restrict__ w1,
    const float* __restrict__ b1,
    const int* __restrict__ cnt, const int* __restrict__ base,
    const int* __restrict__ lists, bf16* __restrict__ hbuf) {
  const int e = blockIdx.z;
  const int n_e = cnt[e];
  const int m0 = blockIdx.x * BM;
  if (m0 >= n_e) return;
  const int n0 = blockIdx.y * BN;
  const int tid = threadIdx.x;

  __shared__ __attribute__((aligned(16))) bf16 As[BM][BK + 8];
  __shared__ __attribute__((aligned(16))) bf16 Bs[BN][BK + 8];  // transposed: [n][k]

  const int lane = tid & 31;
  const int wave = tid >> 5;
  const int wmOff = (wave & 3) * 32;   // 4 waves along M
  const int wnOff = (wave >> 2) * 64;  // 2 waves along N
  const int half = lane >> 4;
  const int ln = lane & 15;

  // A staging: 128 rows x 32 k, 2 threads/row, 16 contiguous k each
  const int arow = tid >> 1;
  const int acol = (tid & 1) * 16;
  const int p_l = m0 + arow;
  const int tok = (p_l < n_e) ? lists[e * MAXTOK + p_l] : -1;
  const float* xrow = x + (size_t)(tok >= 0 ? tok : 0) * H_;
  // B staging: 32 k rows x 128 n, 16 contiguous n per thread
  const int brow = tid >> 3;
  const int bcol = (tid & 7) * 16;
  const float* wmat = w1 + (size_t)e * H_ * I_;

  v8f acc[2][4] = {};

  for (int k0 = 0; k0 < H_; k0 += BK) {
    {  // stage A (fp32 -> bf16)
      bf16* dst = &As[arow][acol];
#pragma unroll
      for (int q = 0; q < 4; ++q) {
        float4 f = *(const float4*)(xrow + k0 + acol + q * 4);
        if (tok < 0) f = make_float4(0.f, 0.f, 0.f, 0.f);
        dst[q * 4 + 0] = (bf16)f.x; dst[q * 4 + 1] = (bf16)f.y;
        dst[q * 4 + 2] = (bf16)f.z; dst[q * 4 + 3] = (bf16)f.w;
      }
    }
    {  // stage B transposed (fp32 -> bf16)
      const float* src = wmat + (size_t)(k0 + brow) * I_ + (n0 + bcol);
#pragma unroll
      for (int j = 0; j < 16; ++j) Bs[bcol + j][brow] = (bf16)src[j];
      if (k0 + BK < H_) __builtin_prefetch(src + (size_t)BK * I_, 0, 1);
    }
    __syncthreads();
    v16bf a[2], b[4];
    a[0] = load_frag(&As[wmOff + ln][half * 8]);
    a[1] = load_frag(&As[wmOff + 16 + ln][half * 8]);
#pragma unroll
    for (int ni = 0; ni < 4; ++ni)
      b[ni] = load_frag(&Bs[wnOff + ni * 16 + ln][half * 8]);
#pragma unroll
    for (int mi = 0; mi < 2; ++mi)
#pragma unroll
      for (int ni = 0; ni < 4; ++ni)
        acc[mi][ni] = __builtin_amdgcn_wmma_f32_16x16x32_bf16(
            false, a[mi], false, b[ni], (short)0, acc[mi][ni], false, false);
    __syncthreads();
  }

  const float* b1e = b1 + (size_t)e * I_;
  const int hb = base[e];
#pragma unroll
  for (int mi = 0; mi < 2; ++mi)
#pragma unroll
    for (int ni = 0; ni < 4; ++ni)
#pragma unroll
      for (int r = 0; r < 8; ++r) {
        int p = m0 + wmOff + mi * 16 + half * 8 + r;
        if (p < n_e) {
          int col = n0 + wnOff + ni * 16 + ln;
          float v = acc[mi][ni][r] + b1e[col];
          hbuf[(size_t)(hb + p) * I_ + col] = (bf16)fmaxf(v, 0.f);
        }
      }
}

// ---------------------------------------------------------------------------
// 4) Expert layer 2: out[tok] += w_e * (h @ w2[e] + b2[e])  (atomic scatter).
//    A (already bf16, contiguous CSR rows) is staged via the Tensor Data
//    Mover: one wave issues TENSOR_LOAD_TO_LDS with a D# describing a 32x128
//    2-D tile with LDS padding (16 DW data + 4 DW pad per row == [BK+8]
//    layout), then waits on TENSORcnt before the workgroup barrier.
// ---------------------------------------------------------------------------
__global__ __launch_bounds__(256) void expert_ffn2(
    const bf16* __restrict__ hbuf, const float* __restrict__ w2,
    const float* __restrict__ b2, const float* __restrict__ wexp,
    const int* __restrict__ cnt, const int* __restrict__ base,
    const int* __restrict__ lists, float* __restrict__ out) {
  const int e = blockIdx.z;
  const int n_e = cnt[e];
  const int m0 = blockIdx.x * BM;
  if (m0 >= n_e) return;
  const int n0 = blockIdx.y * BN;
  const int tid = threadIdx.x;

  __shared__ __attribute__((aligned(16))) bf16 As[BM][BK + 8];
  __shared__ __attribute__((aligned(16))) bf16 Bs[BN][BK + 8];

  const int lane = tid & 31;
  const int wave = tid >> 5;
  const int wmOff = (wave & 3) * 32;
  const int wnOff = (wave >> 2) * 64;
  const int half = lane >> 4;
  const int ln = lane & 15;

  const int brow = tid >> 3;
  const int bcol = (tid & 7) * 16;
  const float* wmat = w2 + (size_t)e * I_ * H_;

  // TDM descriptor invariants (wave-uniform)
  const unsigned lds_a = (unsigned)(uintptr_t)(&As[0][0]);  // LDS byte address
  const int rows_rem = n_e - m0;
  const bf16* tile0 = hbuf + (size_t)(base[e] + m0) * I_;

  v8f acc[2][4] = {};

  for (int k0 = 0; k0 < I_; k0 += BK) {
    if (tid < 32) {  // wave 0: issue TDM load of the 32x128 bf16 A tile
      unsigned long long ga = (unsigned long long)(uintptr_t)(tile0 + k0);
      v4u g0;
      g0[0] = 1u;                                   // count=1, user descriptor
      g0[1] = lds_a;                                // lds_addr
      g0[2] = (unsigned)(ga & 0xFFFFFFFFu);         // global_addr[31:0]
      g0[3] = (unsigned)((ga >> 32) & 0x1FFFFFFu)   // global_addr[56:32]
              | (2u << 30);                         // type=2 ("image")
      v8i_ g1;
      g1[0] = (1 << 16)      // data_size = 2 bytes
            | (1 << 20)      // pad_enable
            | (3 << 22)      // pad_interval: 16 DWORDs (=32 bf16 row)
            | (3 << 25);     // pad_amount: 4 DWORDs (=8 bf16 pad)
      g1[1] = (int)(((unsigned)I_ & 0xFFFFu) << 16);          // tensor_dim0 lo16
      g1[2] = (int)(((unsigned)I_ >> 16) & 0xFFFFu)           // tensor_dim0 hi16
            | (int)(((unsigned)rows_rem & 0xFFFFu) << 16);    // tensor_dim1 lo16
      g1[3] = (int)(((unsigned)rows_rem >> 16) & 0xFFFFu)     // tensor_dim1 hi16
            | (int)(32u << 16);                               // tile_dim0 = BK
      g1[4] = (int)(unsigned)BM;                              // tile_dim1 = 128
      g1[5] = I_;                                             // dim0_stride lo32
      g1[6] = 0;
      g1[7] = 0;
      v4i_ z4 = {};
      v8i_ z8 = {};
      __builtin_amdgcn_tensor_load_to_lds(g0, g1, z4, z4, z8, 0);
    }
    {  // stage B transposed (fp32 -> bf16), all waves
      const float* src = wmat + (size_t)(k0 + brow) * H_ + (n0 + bcol);
#pragma unroll
      for (int j = 0; j < 16; ++j) Bs[bcol + j][brow] = (bf16)src[j];
      if (k0 + BK < I_) __builtin_prefetch(src + (size_t)BK * H_, 0, 1);
    }
    if (tid < 32) __builtin_amdgcn_s_wait_tensorcnt(0);
    __syncthreads();
    v16bf a[2], b[4];
    a[0] = load_frag(&As[wmOff + ln][half * 8]);
    a[1] = load_frag(&As[wmOff + 16 + ln][half * 8]);
#pragma unroll
    for (int ni = 0; ni < 4; ++ni)
      b[ni] = load_frag(&Bs[wnOff + ni * 16 + ln][half * 8]);
#pragma unroll
    for (int mi = 0; mi < 2; ++mi)
#pragma unroll
      for (int ni = 0; ni < 4; ++ni)
        acc[mi][ni] = __builtin_amdgcn_wmma_f32_16x16x32_bf16(
            false, a[mi], false, b[ni], (short)0, acc[mi][ni], false, false);
    __syncthreads();
  }

  const float we = wexp[e];
  const float* b2e = b2 + (size_t)e * H_;
#pragma unroll
  for (int mi = 0; mi < 2; ++mi)
#pragma unroll
    for (int ni = 0; ni < 4; ++ni)
#pragma unroll
      for (int r = 0; r < 8; ++r) {
        int p = m0 + wmOff + mi * 16 + half * 8 + r;
        if (p < n_e) {
          int tok = lists[e * MAXTOK + p];
          int col = n0 + wnOff + ni * 16 + ln;
          atomicAdd(&out[(size_t)tok * H_ + col], we * (acc[mi][ni][r] + b2e[col]));
        }
      }
}

// ---------------------------------------------------------------------------
// 5) Mean-pool over sequence, 6/7) phi MLP, 8) conditional scaling.
// ---------------------------------------------------------------------------
__global__ void pool_kernel(const float* __restrict__ out, float* __restrict__ pooled) {
  int idx = blockIdx.x * 256 + threadIdx.x;  // B*H total
  int b = idx / H_, h = idx % H_;
  float s = 0.f;
  for (int t = 0; t < S_; ++t) s += out[(size_t)(b * S_ + t) * H_ + h];
  pooled[idx] = s * (1.0f / (float)S_);
}

__global__ void phi1_kernel(const float* __restrict__ pooled,
                            const float* __restrict__ Wp1,
                            const float* __restrict__ bp1, float* __restrict__ y1) {
  int b = blockIdx.y;
  int j = blockIdx.x * 256 + threadIdx.x;  // H/2 outputs
  const float* pr = pooled + (size_t)b * H_;
  float s = 0.f;
  for (int h = 0; h < H_; ++h) s += pr[h] * Wp1[(size_t)h * (H_ / 2) + j];
  y1[b * (H_ / 2) + j] = fmaxf(s + bp1[j], 0.f);
}

__global__ void phi2_kernel(const float* __restrict__ y1,
                            const float* __restrict__ Wp2,
                            const float* __restrict__ bp2, float* __restrict__ scale) {
  int b = blockIdx.x, tid = threadIdx.x;
  __shared__ float red[256];
  float s = 0.f;
  for (int j = tid; j < H_ / 2; j += 256) s += y1[b * (H_ / 2) + j] * Wp2[j];
  red[tid] = s;
  __syncthreads();
  for (int off = 128; off > 0; off >>= 1) {
    if (tid < off) red[tid] += red[tid + off];
    __syncthreads();
  }
  if (tid == 0) {
    float phi = 1.f / (1.f + __expf(-(red[0] + bp2[0])));
    scale[b] = (phi > 0.7f) ? 0.2f : 1.0f;  // 1 - 0.8*mask
  }
}

__global__ void scale_kernel(float* __restrict__ out, const float* __restrict__ scale) {
  size_t idx = (size_t)blockIdx.x * 256 + threadIdx.x;
  int b = (int)(idx / ((size_t)S_ * H_));
  out[idx] *= scale[b];
}

// ---------------------------------------------------------------------------
extern "C" void kernel_launch(void* const* d_in, const int* in_sizes, int n_in,
                              void* d_out, int out_size, void* d_ws, size_t ws_size,
                              hipStream_t stream) {
  (void)in_sizes; (void)n_in; (void)ws_size;
  const float* x   = (const float*)d_in[0];
  const float* Wr  = (const float*)d_in[1];
  const float* br  = (const float*)d_in[2];
  const float* w1  = (const float*)d_in[3];
  const float* b1  = (const float*)d_in[4];
  const float* w2  = (const float*)d_in[5];
  const float* b2  = (const float*)d_in[6];
  const float* Wp1 = (const float*)d_in[7];
  const float* bp1 = (const float*)d_in[8];
  const float* Wp2 = (const float*)d_in[9];
  const float* bp2 = (const float*)d_in[10];
  float* out = (float*)d_out;

  char* ws = (char*)d_ws;
  size_t off = 0;
  auto walloc = [&](size_t bytes, size_t align) -> char* {
    off = (off + align - 1) & ~(align - 1);
    char* p = ws + off; off += bytes; return p;
  };
  int*   cnt    = (int*)  walloc(E_ * sizeof(int), 256);
  float* sumw   = (float*)walloc(E_ * sizeof(float), 4);
  float* wexp   = (float*)walloc(E_ * sizeof(float), 4);
  int*   base   = (int*)  walloc(E_ * sizeof(int), 4);
  float* scale  = (float*)walloc(B_ * sizeof(float), 4);
  int*   lists  = (int*)  walloc((size_t)E_ * MAXTOK * sizeof(int), 256);
  float* pooled = (float*)walloc((size_t)B_ * H_ * sizeof(float), 256);
  float* y1     = (float*)walloc((size_t)B_ * (H_ / 2) * sizeof(float), 256);
  bf16*  hbuf   = (bf16*) walloc((size_t)TOTASSIGN * I_ * sizeof(bf16), 256);

  (void)hipMemsetAsync(cnt, 0, E_ * sizeof(int), stream);
  (void)hipMemsetAsync(sumw, 0, E_ * sizeof(float), stream);
  (void)hipMemsetAsync(out, 0, (size_t)out_size * sizeof(float), stream);

  router_kernel<<<NTOK, 256, 0, stream>>>(x, Wr, br, cnt, sumw, lists);
  finalize_kernel<<<1, 32, 0, stream>>>(cnt, sumw, wexp, base);

  dim3 g1(MAXTOK / BM, I_ / BN, E_);  // 32 x 40 x 12, early-exit past cnt[e]
  expert_ffn1<<<g1, 256, 0, stream>>>(x, w1, b1, cnt, base, lists, hbuf);

  dim3 g2(MAXTOK / BM, H_ / BN, E_);  // 32 x 16 x 12
  expert_ffn2<<<g2, 256, 0, stream>>>(hbuf, w2, b2, wexp, cnt, base, lists, out);

  pool_kernel<<<(B_ * H_) / 256, 256, 0, stream>>>(out, pooled);
  dim3 gp1((H_ / 2) / 256, B_);
  phi1_kernel<<<gp1, 256, 0, stream>>>(pooled, Wp1, bp1, y1);
  phi2_kernel<<<B_, 256, 0, stream>>>(y1, Wp2, bp2, scale);
  scale_kernel<<<(B_ * S_ * H_) / 256, 256, 0, stream>>>(out, scale);
}
